// _NonLocalBlockND_Group_17583596110628
// MI455X (gfx1250) — matrix operand order repaired and checked by reading, not compiled, not run on hardware
//
#include <hip/hip_runtime.h>
#include <hip/hip_bf16.h>

// Problem constants (match reference)
#define BSZ 2
#define CCH 256           // C
#define NSP 3136          // H*W = 56*56
#define GRP 8             // G
#define DG  16            // D = CI/G
#define CIC 128           // CI
#define EPSV 1e-5f

typedef __attribute__((ext_vector_type(16))) __bf16 v16bf;
typedef __attribute__((ext_vector_type(8)))  __bf16 v8bf;
typedef __attribute__((ext_vector_type(8)))  float  v8f;

union V16 { v16bf v; v8bf h[2]; };

// ---------------------------------------------------------------------------
// K0: fp32 -> bf16 copy of x, plus a transposed bf16 copy (xT[n][c]).
// ---------------------------------------------------------------------------
__global__ void k_convert(const float* __restrict__ x,
                          __bf16* __restrict__ xb, __bf16* __restrict__ xt) {
  int idx = blockIdx.x * 256 + threadIdx.x;
  if (idx >= BSZ * CCH * NSP) return;
  int n  = idx % NSP;
  int bc = idx / NSP;
  int c  = bc % CCH;
  int b  = bc / CCH;
  float v = x[idx];
  __bf16 bv = (__bf16)v;
  xb[idx] = bv;
  xt[((size_t)b * NSP + n) * CCH + c] = bv;
}

// ---------------------------------------------------------------------------
// K0b: per-channel row sums xs[b][c] = sum_n x[b][c][n] (for the bias terms of S)
// ---------------------------------------------------------------------------
__global__ void k_rowsum(const float* __restrict__ x, float* __restrict__ xs) {
  __shared__ float red[256];
  int bc = blockIdx.x;                       // over B*C
  const float* row = x + (size_t)bc * NSP;
  float s = 0.f;
  for (int n = threadIdx.x; n < NSP; n += 256) s += row[n];
  red[threadIdx.x] = s;
  __syncthreads();
  for (int off = 128; off > 0; off >>= 1) {
    if (threadIdx.x < off) red[threadIdx.x] += red[threadIdx.x + off];
    __syncthreads();
  }
  if (threadIdx.x == 0) xs[bc] = red[0];
}

// ---------------------------------------------------------------------------
// K1: xxT[b] = x_b * x_b^T  (256x256 per batch), bf16 WMMA, fp32 accumulate.
// One wave per 16x16 output tile. A lane l<16 holds row M=l (K 0..7 | 16..23),
// lane l>=16 holds row M=l-16 (K 8..15 | 24..31). B lane l holds column
// N=l&15 with the same K split as A; B's column n is x's row j0+n,
// so both operands are contiguous 16B loads from row-major x_bf16.
// ---------------------------------------------------------------------------
__global__ void k_xxt(const __bf16* __restrict__ xb, float* __restrict__ xxt) {
  const int i0 = blockIdx.x * 16;
  const int j0 = blockIdx.y * 16;
  const int b  = blockIdx.z;
  const int lane = threadIdx.x;          // 0..31
  const int m    = lane & 15;
  const int half = lane >> 4;
  const __bf16* arow = xb + (size_t)(b * CCH + i0 + m) * NSP;
  const __bf16* brow = xb + (size_t)(b * CCH + j0 + m) * NSP;
  const int a0 = half * 8, a1 = 16 + half * 8, b0 = half * 16;
  v8f acc = {};
  for (int k = 0; k < NSP; k += 32) {
    V16 A, Bm;
    A.h[0]  = *(const v8bf*)(arow + k + a0);
    A.h[1]  = *(const v8bf*)(arow + k + a1);
    Bm.h[0] = *(const v8bf*)(brow + k + a0);
    Bm.h[1] = *(const v8bf*)(brow + k + a1);
    acc = __builtin_amdgcn_wmma_f32_16x16x32_bf16(false, A.v, false, Bm.v,
                                                  (short)0, acc, false, false);
  }
#pragma unroll
  for (int r = 0; r < 8; ++r) {          // C/D layout: VGPR r -> M=r (+8 for upper lanes)
    int row = i0 + r + half * 8;
    xxt[((size_t)b * CCH + row) * CCH + j0 + m] = acc[r];
  }
}

// ---------------------------------------------------------------------------
// K2: S[b][g] = (W_phi (x x^T) W_g^T + (W_phi xs) b_g^T + b_phi (W_g xs)^T
//               + N b_phi b_g^T) / N          (16x16 per (b,g))
// ---------------------------------------------------------------------------
__global__ void k_S(const float* __restrict__ xxt, const float* __restrict__ xs,
                    const float* __restrict__ wg, const float* __restrict__ bg,
                    const float* __restrict__ wphi, const float* __restrict__ bphi,
                    float* __restrict__ S) {
  __shared__ float V[16 * 260];          // stride 260 to spread LDS banks
  const int b = blockIdx.x / GRP;
  const int g = blockIdx.x % GRP;
  const int t = threadIdx.x;             // 0..255
  // stage 1: V[dg][i=t] = sum_j xxT[b][i][j] * W_g[g][dg][j]
  {
    const float* xrow = xxt + ((size_t)b * CCH + t) * CCH;
    for (int dg = 0; dg < DG; ++dg) {
      const float* wrow = wg + ((size_t)g * DG + dg) * CCH;
      float s = 0.f;
      for (int j = 0; j < CCH; ++j) s += xrow[j] * wrow[j];
      V[dg * 260 + t] = s;
    }
  }
  __syncthreads();
  // stage 2: one thread per S element
  const int dphi = t >> 4, dg = t & 15;
  const float* wp  = wphi + ((size_t)g * DG + dphi) * CCH;
  const float* wgr = wg   + ((size_t)g * DG + dg)   * CCH;
  const float* xsb = xs + b * CCH;
  float s = 0.f, phx = 0.f, gx = 0.f;
  for (int i = 0; i < CCH; ++i) {
    s   += wp[i]  * V[dg * 260 + i];
    phx += wp[i]  * xsb[i];
    gx  += wgr[i] * xsb[i];
  }
  float bphiv = bphi[g * DG + dphi];
  float bgv   = bg[g * DG + dg];
  float val = (s + phx * bgv + bphiv * gx + (float)NSP * bphiv * bgv)
              * (1.f / (float)NSP);
  S[(((size_t)b * GRP + g) * DG + dphi) * DG + dg] = val;
}

// ---------------------------------------------------------------------------
// K3: TT[b][g*16+d][j] = sum_d' S[b][g][d'][d] * W_theta[g][d'][j]   (CIxC)
//     ybias[b][g*16+d]  = sum_d' S[b][g][d'][d] * b_theta[g][d']
// ---------------------------------------------------------------------------
__global__ void k_TT(const float* __restrict__ S, const float* __restrict__ wtheta,
                     const float* __restrict__ btheta,
                     float* __restrict__ TT, float* __restrict__ ybias) {
  int idx = blockIdx.x * 256 + threadIdx.x;     // over B*CI*C
  if (idx >= BSZ * CIC * CCH) return;
  int j = idx % CCH;
  int k = (idx / CCH) % CIC;
  int b = idx / (CCH * CIC);
  int g = k / DG, d = k % DG;
  const float* Sg = S + ((size_t)b * GRP + g) * DG * DG;
  float s = 0.f;
  for (int dp = 0; dp < DG; ++dp)
    s += Sg[dp * DG + d] * wtheta[((size_t)g * DG + dp) * CCH + j];
  TT[idx] = s;
  if (j == 0) {
    float yb = 0.f;
    for (int dp = 0; dp < DG; ++dp) yb += Sg[dp * DG + d] * btheta[g * DG + dp];
    ybias[b * CIC + k] = yb;
  }
}

// ---------------------------------------------------------------------------
// K4: Mhat[b] = diag(inv) * (W_out @ TT[b])  -> bf16 (256x256 per batch)
//     chat[b][i] = inv[i]*(b_out[i] + W_out[i,:]·ybias[b]) + (beta - mean*inv)[i]
// ---------------------------------------------------------------------------
__global__ void k_M(const float* __restrict__ TT, const float* __restrict__ ybias,
                    const float* __restrict__ wout, const float* __restrict__ bout,
                    const float* __restrict__ gamma, const float* __restrict__ beta,
                    const float* __restrict__ rmean, const float* __restrict__ rvar,
                    __bf16* __restrict__ Mhat, float* __restrict__ chat) {
  int idx = blockIdx.x * 256 + threadIdx.x;     // over B*C*C
  if (idx >= BSZ * CCH * CCH) return;
  int j = idx % CCH;
  int i = (idx / CCH) % CCH;
  int b = idx / (CCH * CCH);
  float inv = gamma[i] * rsqrtf(rvar[i] + EPSV);
  const float* wrow = wout + (size_t)i * CIC;
  const float* tcol = TT + (size_t)b * CIC * CCH + j;
  float acc = 0.f;
  for (int k = 0; k < CIC; ++k) acc += wrow[k] * tcol[(size_t)k * CCH];
  Mhat[idx] = (__bf16)(inv * acc);
  if (j == 0) {
    float c = bout[i];
    const float* yb = ybias + b * CIC;
    for (int k = 0; k < CIC; ++k) c += wrow[k] * yb[k];
    chat[b * CCH + i] = inv * c + (beta[i] - rmean[i] * inv);
  }
}

// ---------------------------------------------------------------------------
// K5: z = relu(Mhat @ x + chat + x). bf16 WMMA, fp32 epilogue (residual in fp32).
// 4 waves/block, each wave owns one 16x16 tile; B operand columns come from
// the transposed copy xT (contiguous 16B loads per lane).
// ---------------------------------------------------------------------------
__global__ void k_out(const __bf16* __restrict__ Mhat, const __bf16* __restrict__ xt,
                      const float* __restrict__ chat, const float* __restrict__ x,
                      float* __restrict__ out) {
  const int b  = blockIdx.z;
  const int i0 = blockIdx.x * 16;
  const int w  = threadIdx.x >> 5;
  const int n0 = (blockIdx.y * 4 + w) * 16;
  const int lane = threadIdx.x & 31;
  const int m    = lane & 15;
  const int half = lane >> 4;
  const __bf16* arow = Mhat + (size_t)(b * CCH + i0 + m) * CCH;
  const __bf16* brow = xt   + ((size_t)b * NSP + n0 + m) * CCH;
  const int a0 = half * 8, a1 = 16 + half * 8, b0 = half * 16;
  v8f acc = {};
  for (int k = 0; k < CCH; k += 32) {
    V16 A, Bm;
    A.h[0]  = *(const v8bf*)(arow + k + a0);
    A.h[1]  = *(const v8bf*)(arow + k + a1);
    Bm.h[0] = *(const v8bf*)(brow + k + a0);
    Bm.h[1] = *(const v8bf*)(brow + k + a1);
    acc = __builtin_amdgcn_wmma_f32_16x16x32_bf16(false, A.v, false, Bm.v,
                                                  (short)0, acc, false, false);
  }
#pragma unroll
  for (int r = 0; r < 8; ++r) {
    int row = i0 + r + half * 8;
    int col = n0 + m;
    size_t o = ((size_t)b * CCH + row) * NSP + col;
    out[o] = fmaxf(acc[r] + chat[b * CCH + row] + x[o], 0.f);
  }
}

// ---------------------------------------------------------------------------
extern "C" void kernel_launch(void* const* d_in, const int* in_sizes, int n_in,
                              void* d_out, int out_size, void* d_ws, size_t ws_size,
                              hipStream_t stream) {
  (void)in_sizes; (void)n_in; (void)out_size; (void)ws_size;
  const float* x     = (const float*)d_in[0];
  const float* w_g   = (const float*)d_in[1];
  const float* b_g   = (const float*)d_in[2];
  const float* w_th  = (const float*)d_in[3];
  const float* b_th  = (const float*)d_in[4];
  const float* w_phi = (const float*)d_in[5];
  const float* b_phi = (const float*)d_in[6];
  const float* w_out = (const float*)d_in[7];
  const float* b_out = (const float*)d_in[8];
  const float* gamma = (const float*)d_in[9];
  const float* beta  = (const float*)d_in[10];
  const float* rmean = (const float*)d_in[11];
  const float* rvar  = (const float*)d_in[12];
  float* out = (float*)d_out;

  char* ws = (char*)d_ws;
  size_t off = 0;
  auto alloc = [&](size_t bytes) -> void* {
    size_t cur = (off + 255) & ~(size_t)255;
    off = cur + bytes;
    return (void*)(ws + cur);
  };
  __bf16* xb    = (__bf16*)alloc((size_t)BSZ * CCH * NSP * 2);
  __bf16* xt    = (__bf16*)alloc((size_t)BSZ * NSP * CCH * 2);
  float*  xs    = (float*) alloc((size_t)BSZ * CCH * 4);
  float*  xxt   = (float*) alloc((size_t)BSZ * CCH * CCH * 4);
  float*  S     = (float*) alloc((size_t)BSZ * GRP * DG * DG * 4);
  float*  TT    = (float*) alloc((size_t)BSZ * CIC * CCH * 4);
  float*  ybias = (float*) alloc((size_t)BSZ * CIC * 4);
  __bf16* Mhat  = (__bf16*)alloc((size_t)BSZ * CCH * CCH * 2);
  float*  chat  = (float*) alloc((size_t)BSZ * CCH * 4);

  const int tot = BSZ * CCH * NSP;
  k_convert<<<(tot + 255) / 256, 256, 0, stream>>>(x, xb, xt);
  k_rowsum<<<BSZ * CCH, 256, 0, stream>>>(x, xs);
  k_xxt<<<dim3(CCH / 16, CCH / 16, BSZ), 32, 0, stream>>>(xb, xxt);
  k_S<<<BSZ * GRP, 256, 0, stream>>>(xxt, xs, w_g, b_g, w_phi, b_phi, S);
  k_TT<<<(BSZ * CIC * CCH + 255) / 256, 256, 0, stream>>>(S, w_th, b_th, TT, ybias);
  k_M<<<(BSZ * CCH * CCH + 255) / 256, 256, 0, stream>>>(TT, ybias, w_out, b_out,
                                                         gamma, beta, rmean, rvar,
                                                         Mhat, chat);
  k_out<<<dim3(CCH / 16, NSP / 64, BSZ), 128, 0, stream>>>(Mhat, xt, chat, x, out);
}
